// MinCutNet_90374701842971
// MI455X (gfx1250) — matrix operand order repaired and checked by reading, not compiled, not run on hardware
//
#include <hip/hip_runtime.h>
#include <hip/hip_bf16.h>

typedef float  v2f   __attribute__((ext_vector_type(2)));
typedef float  v8f   __attribute__((ext_vector_type(8)));
typedef __bf16 v16bf __attribute__((ext_vector_type(16)));

#if defined(__AMDGCN__) && __has_builtin(__builtin_amdgcn_wmma_f32_16x16x4_f32)
#define USE_F32_WMMA 1
#else
#define USE_F32_WMMA 0
#endif

#define FH 256   // hidden width
#define KC 64    // cluster count

// ---------------- init: deg=1 (self loop weight), drow=0, ss/red=0 ----------
__global__ void k_init(float* deg, float* drow, float* ssred, int N) {
  int i = blockIdx.x * blockDim.x + threadIdx.x;
  int stride = gridDim.x * blockDim.x;
  for (int j = i; j < N; j += stride) { deg[j] = 1.0f; drow[j] = 0.0f; }
  for (int j = i; j < 64 * 64 + 16; j += stride) ssred[j] = 0.0f;
}

// ---------------- degree accumulation --------------------------------------
__global__ void k_deg(const int* __restrict__ row, const int* __restrict__ col,
                      const float* __restrict__ ew, float* deg, float* drow, int E) {
  int e = blockIdx.x * blockDim.x + threadIdx.x;
  if (e < E) {
    float wv = ew[e];
    atomicAdd(deg + col[e], wv);
    atomicAdd(drow + row[e], wv);
  }
}

__global__ void k_dinv(const float* __restrict__ deg, float* dinv, int N) {
  int i = blockIdx.x * blockDim.x + threadIdx.x;
  if (i < N) { float d = deg[i]; dinv[i] = d > 0.0f ? rsqrtf(d) : 0.0f; }
}

__global__ void k_norm(const int* __restrict__ row, const int* __restrict__ col,
                       const float* __restrict__ ew, const float* __restrict__ dinv,
                       float* nw, int E) {
  int e = blockIdx.x * blockDim.x + threadIdx.x;
  if (e < E) nw[e] = dinv[row[e]] * ew[e] * dinv[col[e]];
}

// ---------------- WMMA GEMM: C[M x Nc] = A[M x K] @ W[K x Nc] ---------------
// One 16x16 output tile per wave; M % 16 == 0, K % 32 == 0, Nc % 16 == 0.
__global__ __launch_bounds__(256) void k_gemm_wmma(
    const float* __restrict__ A, const float* __restrict__ W,
    float* __restrict__ C, int M, int K, int Nc) {
  const int lane = threadIdx.x & 31;
  const int wave = threadIdx.x >> 5;
  const int tilesN = Nc >> 4;
  const int tiles = (M >> 4) * tilesN;
  const int tid = blockIdx.x * 8 + wave;   // wave-uniform
  if (tid >= tiles) return;
  const int tm = tid / tilesN;
  const int tn = tid - tm * tilesN;
  const int l15 = lane & 15;
  const int g = lane >> 4;

  v8f c = {};
  const float* arow = A + (size_t)(tm * 16 + l15) * K;
  const float* bcol = W + tn * 16 + l15;

#if USE_F32_WMMA
  for (int k0 = 0; k0 < K; k0 += 4) {
    const int kb = k0 + 2 * g;          // VGPR0: K=0|2 ; VGPR1: K=1|3
    v2f a, b;
    a.x = arow[kb];
    a.y = arow[kb + 1];
    b.x = bcol[(size_t)kb * Nc];
    b.y = bcol[(size_t)(kb + 1) * Nc];
    c = __builtin_amdgcn_wmma_f32_16x16x4_f32(false, a, false, b, (short)0, c,
                                              false, false);
  }
#else
  for (int k0 = 0; k0 < K; k0 += 32) {
    v16bf a, b;
#pragma unroll
    for (int e2 = 0; e2 < 16; ++e2) {
      // A 16x32 bf16 layout: VGPR0-3 hold K = g*8 + {0..7}; VGPR4-7 hold +16
      int ka = (e2 < 8) ? (g * 8 + e2) : (16 + g * 8 + (e2 - 8));
      a[e2] = (__bf16)arow[k0 + ka];
      // B 32x16 bf16 layout: same K order as A (g*8 + {0..7}; +16 for e2>=8)
      int kb = ka;
      b[e2] = (__bf16)bcol[(size_t)(k0 + kb) * Nc];
    }
    c = __builtin_amdgcn_wmma_f32_16x16x32_bf16(false, a, false, b, (short)0, c,
                                                false, false);
  }
#endif
  // C/D layout: VGPR r holds M = r (lanes 0-15) / M = 8+r (lanes 16-31)
  float* crow = C + (size_t)(tm * 16 + g * 8) * Nc + tn * 16 + l15;
#pragma unroll
  for (int r = 0; r < 8; ++r) crow[(size_t)r * Nc] = c[r];
}

// ---------------- self-loop term: out = dinv[n]^2 * hW ----------------------
__global__ void k_selfloop(const float* __restrict__ hW, const float* __restrict__ dinv,
                           float* out, int N) {
  int i = blockIdx.x * blockDim.x + threadIdx.x;
  if (i < N * FH) {
    int n = i >> 8;   // /256
    float d = dinv[n];
    out[i] = d * d * hW[i];
  }
}

// ---------------- edge scatter-add: out[col] += norm * hW[row] --------------
__global__ void k_scatter(const float* __restrict__ hW, const float* __restrict__ nw,
                          const int* __restrict__ row, const int* __restrict__ col,
                          float* out, int E) {
  int t = blockIdx.x * blockDim.x + threadIdx.x;
  if (t >= E * 64) return;
  int e = t >> 6;
  int f0 = (t & 63) * 4;
  float wv = nw[e];
  size_t rb = (size_t)row[e] * FH + f0;
  size_t cb = (size_t)col[e] * FH + f0;
  float4 h = *(const float4*)(hW + rb);
  atomicAdd(out + cb + 0, wv * h.x);
  atomicAdd(out + cb + 1, wv * h.y);
  atomicAdd(out + cb + 2, wv * h.z);
  atomicAdd(out + cb + 3, wv * h.w);
}

__global__ void k_bias_relu(const float* __restrict__ acc, const float* __restrict__ b,
                            float* h, int N) {
  int i = blockIdx.x * blockDim.x + threadIdx.x;
  if (i < N * FH) {
    float v = acc[i] + b[i & (FH - 1)];
    h[i] = v > 0.0f ? v : 0.0f;
  }
}

// ---------------- softmax over K=64 per row ---------------------------------
__global__ void k_softmax(const float* __restrict__ logits, const float* __restrict__ bp,
                          float* s, int N) {
  int i = blockIdx.x * blockDim.x + threadIdx.x;
  if (i >= N) return;
  const float* L = logits + (size_t)i * KC;
  float* S = s + (size_t)i * KC;
  float m = -1e30f;
  for (int j = 0; j < KC; ++j) { float v = L[j] + bp[j]; m = v > m ? v : m; }
  float sum = 0.0f;
  for (int j = 0; j < KC; ++j) { float v = __expf(L[j] + bp[j] - m); S[j] = v; sum += v; }
  float inv = 1.0f / sum;
  for (int j = 0; j < KC; ++j) S[j] *= inv;
}

// ---------------- mincut numerator: sum_e ew * dot(s[row], s[col]) ----------
__global__ __launch_bounds__(256) void k_mincut_num(
    const float* __restrict__ sm, const int* __restrict__ row, const int* __restrict__ col,
    const float* __restrict__ ew, float* red, int E) {
  int e = blockIdx.x * blockDim.x + threadIdx.x;
  float acc = 0.0f;
  if (e < E) {
    const float* a = sm + (size_t)row[e] * KC;
    const float* b = sm + (size_t)col[e] * KC;
    float d = 0.0f;
    for (int j = 0; j < KC; j += 4) {
      float4 x = *(const float4*)(a + j);
      float4 y = *(const float4*)(b + j);
      d += x.x * y.x + x.y * y.y + x.z * y.z + x.w * y.w;
    }
    acc = ew[e] * d;
  }
  __shared__ float r[256];
  r[threadIdx.x] = acc;
  __syncthreads();
  for (int s2 = 128; s2 > 0; s2 >>= 1) {
    if (threadIdx.x < s2) r[threadIdx.x] += r[threadIdx.x + s2];
    __syncthreads();
  }
  if (threadIdx.x == 0) atomicAdd(red + 0, r[0]);
}

// ---------------- mincut denominator: sum_i drow[i] * ||s_i||^2 -------------
__global__ __launch_bounds__(256) void k_mincut_den(
    const float* __restrict__ sm, const float* __restrict__ drow, float* red, int N) {
  int i = blockIdx.x * blockDim.x + threadIdx.x;
  float acc = 0.0f;
  if (i < N) {
    const float* a = sm + (size_t)i * KC;
    float s2 = 0.0f;
    for (int j = 0; j < KC; j += 4) {
      float4 x = *(const float4*)(a + j);
      s2 += x.x * x.x + x.y * x.y + x.z * x.z + x.w * x.w;
    }
    acc = drow[i] * s2;
  }
  __shared__ float r[256];
  r[threadIdx.x] = acc;
  __syncthreads();
  for (int s2 = 128; s2 > 0; s2 >>= 1) {
    if (threadIdx.x < s2) r[threadIdx.x] += r[threadIdx.x + s2];
    __syncthreads();
  }
  if (threadIdx.x == 0) atomicAdd(red + 1, r[0]);
}

// ---------------- ss = s^T s (64x64), per-block LDS accumulators ------------
__global__ __launch_bounds__(256) void k_sts(const float* __restrict__ sm, float* ss,
                                             int N, int rowsPerBlock) {
  __shared__ float acc[KC * KC];
  __shared__ float srow[KC];
  for (int e = threadIdx.x; e < KC * KC; e += 256) acc[e] = 0.0f;
  int r0 = blockIdx.x * rowsPerBlock;
  int r1 = r0 + rowsPerBlock;
  if (r1 > N) r1 = N;
  __syncthreads();
  for (int rr = r0; rr < r1; ++rr) {
    if (threadIdx.x < KC) srow[threadIdx.x] = sm[(size_t)rr * KC + threadIdx.x];
    __syncthreads();
#pragma unroll
    for (int q = 0; q < 16; ++q) {
      int e = threadIdx.x * 16 + q;
      acc[e] += srow[e >> 6] * srow[e & 63];
    }
    __syncthreads();
  }
  for (int e = threadIdx.x; e < KC * KC; e += 256) atomicAdd(ss + e, acc[e]);
}

// ---------------- finalize: ortho loss + combine ----------------------------
__global__ __launch_bounds__(256) void k_finalize(const float* __restrict__ ss,
                                                  const float* __restrict__ red,
                                                  float* out) {
  __shared__ float r[256];
  __shared__ float fro_s;
  int t = threadIdx.x;
  float a = 0.0f;
  for (int e = t; e < 4096; e += 256) { float v = ss[e]; a += v * v; }
  r[t] = a;
  __syncthreads();
  for (int s2 = 128; s2 > 0; s2 >>= 1) {
    if (t < s2) r[t] += r[t + s2];
    __syncthreads();
  }
  if (t == 0) fro_s = sqrtf(r[0]);
  __syncthreads();
  float inv = fro_s > 0.0f ? 1.0f / fro_s : 0.0f;
  float b = 0.0f;
  for (int e = t; e < 4096; e += 256) {
    float v = ss[e] * inv - (((e % 65) == 0) ? 0.125f : 0.0f);  // I/sqrt(64)
    b += v * v;
  }
  __syncthreads();
  r[t] = b;
  __syncthreads();
  for (int s2 = 128; s2 > 0; s2 >>= 1) {
    if (t < s2) r[t] += r[t + s2];
    __syncthreads();
  }
  if (t == 0) out[0] = -(red[0] / red[1]) + sqrtf(r[0]);
}

// ---------------------------------------------------------------------------
extern "C" void kernel_launch(void* const* d_in, const int* in_sizes, int n_in,
                              void* d_out, int out_size, void* d_ws, size_t ws_size,
                              hipStream_t stream) {
  (void)n_in; (void)out_size; (void)ws_size;
  const float* x  = (const float*)d_in[0];
  const int*   ei = (const int*)d_in[1];
  const float* ew = (const float*)d_in[2];
  const float* W1 = (const float*)d_in[3];
  const float* b1 = (const float*)d_in[4];
  const float* W2 = (const float*)d_in[5];
  const float* b2 = (const float*)d_in[6];
  const float* Wp = (const float*)d_in[7];
  const float* bp = (const float*)d_in[8];
  float* out = (float*)d_out;

  const int N = in_sizes[0] / 128;   // 10000
  const int E = in_sizes[2];         // 320000
  const int* row = ei;
  const int* col = ei + E;

  float* w = (float*)d_ws;
  size_t off = 0;
  auto take = [&](size_t n) -> float* {
    float* p = w + off;
    off += (n + 63) & ~(size_t)63;   // 256B-align every buffer
    return p;
  };
  float* deg   = take((size_t)N);
  float* drow  = take((size_t)N);
  float* dinv  = take((size_t)N);
  float* normw = take((size_t)E);
  float* bufA  = take((size_t)N * FH);
  float* bufB  = take((size_t)N * FH);
  float* hbuf  = take((size_t)N * FH);
  float* ss    = take((size_t)64 * 64);  // red follows contiguously
  float* red   = take((size_t)16);
  (void)red;  // red == ss + 4096 by construction

  const int TPB = 256;
  const int gbN = (N + TPB - 1) / TPB;
  const int gbE = (E + TPB - 1) / TPB;

  k_init<<<gbN, TPB, 0, stream>>>(deg, drow, ss, N);
  k_deg<<<gbE, TPB, 0, stream>>>(row, col, ew, deg, drow, E);
  k_dinv<<<gbN, TPB, 0, stream>>>(deg, dinv, N);
  k_norm<<<gbE, TPB, 0, stream>>>(row, col, ew, dinv, normw, E);

  const int gbNF = (N * FH + TPB - 1) / TPB;
  const int gbS  = (E * 64 + TPB - 1) / TPB;

  // ---- GCN layer 1: h = relu(Ahat @ (x @ W1) + b1) ----
  {
    int tiles = (N / 16) * (FH / 16);
    k_gemm_wmma<<<(tiles + 7) / 8, 256, 0, stream>>>(x, W1, bufA, N, 128, FH);
    k_selfloop<<<gbNF, TPB, 0, stream>>>(bufA, dinv, bufB, N);
    k_scatter<<<gbS, TPB, 0, stream>>>(bufA, normw, row, col, bufB, E);
    k_bias_relu<<<gbNF, TPB, 0, stream>>>(bufB, b1, hbuf, N);
  }
  // ---- GCN layer 2 ----
  {
    int tiles = (N / 16) * (FH / 16);
    k_gemm_wmma<<<(tiles + 7) / 8, 256, 0, stream>>>(hbuf, W2, bufA, N, FH, FH);
    k_selfloop<<<gbNF, TPB, 0, stream>>>(bufA, dinv, bufB, N);
    k_scatter<<<gbS, TPB, 0, stream>>>(bufA, normw, row, col, bufB, E);
    k_bias_relu<<<gbNF, TPB, 0, stream>>>(bufB, b2, hbuf, N);
  }
  // ---- pooling: s = softmax(h @ Wp + bp) ----
  {
    int tiles = (N / 16) * (KC / 16);
    k_gemm_wmma<<<(tiles + 7) / 8, 256, 0, stream>>>(hbuf, Wp, bufA, N, FH, KC);
    k_softmax<<<gbN, TPB, 0, stream>>>(bufA, bp, bufB, N);   // bufB = s
  }
  // ---- losses ----
  k_mincut_num<<<gbE, TPB, 0, stream>>>(bufB, row, col, ew, red, E);
  k_mincut_den<<<gbN, TPB, 0, stream>>>(bufB, drow, red, N);
  const int nb = 80;
  k_sts<<<nb, 256, 0, stream>>>(bufB, ss, N, (N + nb - 1) / nb);
  k_finalize<<<1, 256, 0, stream>>>(ss, red, out);
}
